// Attention_39384850104955
// MI455X (gfx1250) — matrix operand order repaired and checked by reading, not compiled, not run on hardware
//
#include <hip/hip_runtime.h>
#include <hip/hip_bf16.h>

typedef __attribute__((ext_vector_type(16))) __bf16 bf16x16;
typedef __attribute__((ext_vector_type(8)))  __bf16 bf16x8;
typedef __attribute__((ext_vector_type(8)))  float  f32x8;
typedef __attribute__((ext_vector_type(4)))  unsigned int u32x4;
typedef __attribute__((ext_vector_type(8)))  int i32x8;
typedef __attribute__((ext_vector_type(4)))  int i32x4;

#if __has_builtin(__builtin_amdgcn_tensor_load_to_lds) && __has_builtin(__builtin_amdgcn_s_wait_tensorcnt)
#define USE_TDM 1
#else
#define USE_TDM 0
#endif

namespace {

constexpr int kB  = 2;
constexpr int kS  = 2048;
constexpr int kD  = 1024;
constexpr int kH  = 16;
constexpr int kDH = 64;
constexpr int kM  = kB * kS;   // 4096 rows for the projection GEMMs

__device__ __forceinline__ unsigned short f32_to_bf16_rne(float f) {
  unsigned int u = __float_as_uint(f);
  u += 0x7FFFu + ((u >> 16) & 1u);
  return (unsigned short)(u >> 16);
}

// A-operand fragment (16-bit, 16xK tile slice of 32 K):
// lane<16: row=lane, K={0..7,16..23}; lane>=16: row=lane-16, K={8..15,24..31}
// caller passes p = &A[row*ld + kbase + (lane>=16 ? 8 : 0)]
__device__ __forceinline__ bf16x16 load_fragA(const unsigned short* p) {
  bf16x8 lo = *(const bf16x8*)(p);
  bf16x8 hi = *(const bf16x8*)(p + 16);
  return __builtin_shufflevector(lo, hi, 0,1,2,3,4,5,6,7,8,9,10,11,12,13,14,15);
}

// B-operand fragment (16-bit, 32x16): lane<16 holds K={0..7,16..23}, lane>=16 holds K={8..15,24..31}
// for column N=lane%16. We store B^T row-major (same K order as the A fragment).
// caller passes p = &BT[n*ld + (lane>=16 ? 8 : 0)]
__device__ __forceinline__ bf16x16 load_fragB(const unsigned short* p) {
  bf16x8 lo = *(const bf16x8*)(p);
  bf16x8 hi = *(const bf16x8*)(p + 16);
  return __builtin_shufflevector(lo, hi, 0,1,2,3,4,5,6,7,8,9,10,11,12,13,14,15);
}

__device__ __forceinline__ float redmax16(float v) {
  v = fmaxf(v, __shfl_xor(v, 1, 32));
  v = fmaxf(v, __shfl_xor(v, 2, 32));
  v = fmaxf(v, __shfl_xor(v, 4, 32));
  v = fmaxf(v, __shfl_xor(v, 8, 32));
  return v;
}
__device__ __forceinline__ float redsum16(float v) {
  v += __shfl_xor(v, 1, 32);
  v += __shfl_xor(v, 2, 32);
  v += __shfl_xor(v, 4, 32);
  v += __shfl_xor(v, 8, 32);
  return v;
}

#if USE_TDM
// Tensor Data Mover: DMA one [rows x 32] bf16 tile (row stride `ld` elements in
// global) into LDS at byte offset lds_off, inserting 4 DWORDs of padding after
// every 16 DWORDs (=> 80B LDS row stride, bank-conflict-free b128 reads).
// D# packing per CDNA5 ISA ch.8 (group0: count/lds/global/type, group1: dims).
__device__ __forceinline__ void tdm_load_tile(unsigned int lds_off,
                                              const unsigned short* gptr,
                                              int ld, int rows) {
  unsigned long long ga = (unsigned long long)(uintptr_t)gptr;
  u32x4 g0;
  g0[0] = 1u;                                             // count=1, user mode
  g0[1] = lds_off;                                        // lds_addr (bytes)
  g0[2] = (unsigned int)(ga & 0xFFFFFFFFull);             // global_addr[31:0]
  g0[3] = (unsigned int)((ga >> 32) & 0x1FFFFFFull)       // global_addr[56:32]
        | (2u << 30);                                     // type=2 ("image")
  i32x8 g1;
  g1[0] = (1 << 16)       // data_size = 1 (2 bytes)
        | (1 << 20)       // pad_enable
        | (3 << 22)       // pad_interval: 16 DWORDs
        | (3 << 25);      // pad_amount:   4 DWORDs
  g1[1] = (int)(((unsigned)ld & 0xFFFFu) << 16);          // tensor_dim0[15:0]
  g1[2] = (int)(((unsigned)ld >> 16) | (0xFFFFu << 16));  // dim0 hi | tensor_dim1=65535
  g1[3] = (int)(32u << 16);                               // tile_dim0 = 32 elements
  g1[4] = rows;                                           // tile_dim1, tile_dim2=0
  g1[5] = ld;                                             // tensor_dim0_stride lo
  g1[6] = 0;
  g1[7] = 0;
  i32x4 z4 = {0, 0, 0, 0};                                // 2D: groups 2/3 unused
  i32x8 z8 = {0, 0, 0, 0, 0, 0, 0, 0};
  // 6-arg form (this toolchain): (g0, g1, g2, g3, extra, cpol)
  __builtin_amdgcn_tensor_load_to_lds(g0, g1, z4, z4, z8, 0);
}
#endif

__global__ void cvt_f32_to_bf16(const float* __restrict__ src,
                                unsigned short* __restrict__ dst, int n4) {
  int i = blockIdx.x * blockDim.x + threadIdx.x;
  if (i < n4) {
    float4 f = ((const float4*)src)[i];
    ushort4 o;
    o.x = f32_to_bf16_rne(f.x);
    o.y = f32_to_bf16_rne(f.y);
    o.z = f32_to_bf16_rne(f.z);
    o.w = f32_to_bf16_rne(f.w);
    ((ushort4*)dst)[i] = o;
  }
}

// C = A[MxK] * W^T, W stored [NxK] row-major (so W rows are B^T rows).
// MODE 0: f32 out, row-major [MxN]
// MODE 1: bf16 out, head-split   q[((b*H+h)*S+s)*DH+dh]
// MODE 2: bf16 out, head-split-T v[((b*H+h)*DH+dh)*S+s]
template <int MODE>
__global__ __launch_bounds__(256)
void gemm_bf16_wmma(const unsigned short* __restrict__ A,
                    const unsigned short* __restrict__ W,
                    void* out, int M, int N, int K) {
  constexpr int BM = 128, BN = 128, BK = 32, LDT = 40;  // 80B rows: conflict-free
  __shared__ __align__(16) unsigned short As[BM * LDT];
  __shared__ __align__(16) unsigned short Bs[BN * LDT];

  const int tid  = threadIdx.x;
  const int wid  = tid >> 5;
  const int lane = tid & 31;
  const int m0 = blockIdx.y * BM;
  const int n0 = blockIdx.x * BN;
  const int wm = (wid >> 2) * 64;   // wave tile: 64(M) x 32(N)
  const int wn = (wid & 3) * 32;

  const int r  = lane & 15;
  const int kf = (lane >> 4) * 8;    // A-frag K base adjust
  const int kb = (lane >> 4) * 8;    // B-frag K base adjust
  const int hl = lane >> 4;

  const f32x8 vzero = {0.f,0.f,0.f,0.f,0.f,0.f,0.f,0.f};
  f32x8 acc[4][2];
#pragma unroll
  for (int i = 0; i < 4; ++i)
#pragma unroll
    for (int j = 0; j < 2; ++j) acc[i][j] = vzero;

#if USE_TDM
  const unsigned int As_off = (unsigned int)(uintptr_t)&As[0];
  const unsigned int Bs_off = (unsigned int)(uintptr_t)&Bs[0];
#else
  const int lr = tid >> 1;           // 0..127 : tile row this thread stages
  const int lc = (tid & 1) * 16;     // 0 or 16: first of its 16 bf16 elements
#endif

  for (int k0 = 0; k0 < K; k0 += BK) {
#if USE_TDM
    // Tensor Data Mover stages both tiles (pads rows 64B->80B in-flight).
    if (wid == 0) {
      tdm_load_tile(As_off, A + (size_t)m0 * K + k0, K, BM);
      __builtin_amdgcn_s_wait_tensorcnt(0);
    } else if (wid == 1) {
      tdm_load_tile(Bs_off, W + (size_t)n0 * K + k0, K, BN);
      __builtin_amdgcn_s_wait_tensorcnt(0);
    }
    __syncthreads();
#else
    const uint4* ga = (const uint4*)(A + (size_t)(m0 + lr) * K + k0 + lc);
    const uint4* gb = (const uint4*)(W + (size_t)(n0 + lr) * K + k0 + lc);
    uint4 a0 = ga[0], a1 = ga[1];
    uint4 b0 = gb[0], b1 = gb[1];
    *(uint4*)(&As[lr * LDT + lc])     = a0;
    *(uint4*)(&As[lr * LDT + lc + 8]) = a1;
    *(uint4*)(&Bs[lr * LDT + lc])     = b0;
    *(uint4*)(&Bs[lr * LDT + lc + 8]) = b1;
    __syncthreads();
#endif

    bf16x16 af[4], bfr[2];
#pragma unroll
    for (int i = 0; i < 4; ++i)
      af[i] = load_fragA(&As[(wm + i * 16 + r) * LDT + kf]);
#pragma unroll
    for (int j = 0; j < 2; ++j)
      bfr[j] = load_fragB(&Bs[(wn + j * 16 + r) * LDT + kb]);
#pragma unroll
    for (int i = 0; i < 4; ++i)
#pragma unroll
      for (int j = 0; j < 2; ++j)
        acc[i][j] = __builtin_amdgcn_wmma_f32_16x16x32_bf16(
            false, af[i], false, bfr[j], (short)0, acc[i][j], false, false);
    __syncthreads();
  }

  const int nn = lane & 15;
#pragma unroll
  for (int i = 0; i < 4; ++i)
#pragma unroll
    for (int j = 0; j < 2; ++j)
#pragma unroll
      for (int jj = 0; jj < 8; ++jj) {
        int gm = m0 + wm + i * 16 + jj + 8 * hl;  // C frag: vgpr jj -> M=jj+8*(lane/16)
        int gc = n0 + wn + j * 16 + nn;
        float v = acc[i][j][jj];
        if (MODE == 0) {
          ((float*)out)[(size_t)gm * N + gc] = v;
        } else {
          int b = gm / kS, s = gm % kS;
          int h = gc / kDH, dh = gc % kDH;
          unsigned short bv = f32_to_bf16_rne(v);
          if (MODE == 1)
            ((unsigned short*)out)[(((size_t)(b * kH + h)) * kS + s) * kDH + dh] = bv;
          else
            ((unsigned short*)out)[(((size_t)(b * kH + h)) * kDH + dh) * kS + s] = bv;
        }
      }
}

// Flash attention: Q[B,H,S,DH] bf16, K[B,H,S,DH] bf16, Vt[B,H,DH,S] bf16,
// bias[1,H,S,S] f32  ->  ctx[B,S,D] bf16. One wave per 16 q rows, kv tiles of 32.
__global__ __launch_bounds__(256)
void attn_fwd(const unsigned short* __restrict__ Q,
              const unsigned short* __restrict__ Kk,
              const unsigned short* __restrict__ Vt,
              const float* __restrict__ bias,
              unsigned short* __restrict__ ctx) {
  constexpr int LDP = 40;
  __shared__ __align__(16) unsigned short Psh[8 * 16 * LDP];

  const int wid  = threadIdx.x >> 5;
  const int lane = threadIdx.x & 31;
  const int b = blockIdx.z, h = blockIdx.y;
  const int bh = b * kH + h;
  const int q0 = blockIdx.x * 128 + wid * 16;

  const float sc = 0.125f * 1.44269504088896340736f;  // 1/sqrt(DH) * log2(e)
  const float bl = 1.44269504088896340736f;           // log2(e) for bias

  const int r  = lane & 15;
  const int kf = (lane >> 4) * 8;
  const int kb = (lane >> 4) * 8;
  const int hl = lane >> 4;
  const int nn = lane & 15;

  // Resident Q A-fragments (K = DH = 64 -> 2 steps of 32)
  bf16x16 aq[2];
#pragma unroll
  for (int t = 0; t < 2; ++t)
    aq[t] = load_fragA(Q + ((size_t)bh * kS + q0 + r) * kDH + t * 32 + kf);

  const f32x8 vzero = {0.f,0.f,0.f,0.f,0.f,0.f,0.f,0.f};
  f32x8 acc[4];
#pragma unroll
  for (int f = 0; f < 4; ++f) acc[f] = vzero;
  float rm[8], rl[8];
#pragma unroll
  for (int j = 0; j < 8; ++j) { rm[j] = -3.0e38f; rl[j] = 0.f; }

  unsigned short* pw = &Psh[wid * 16 * LDP];

  for (int kv0 = 0; kv0 < kS; kv0 += 32) {
    // ---- S = Q * K^T  (two 16-col kv fragments) ----
    f32x8 s0 = vzero, s1 = vzero;
#pragma unroll
    for (int t = 0; t < 2; ++t) {
      bf16x16 bk0 = load_fragB(Kk + ((size_t)bh * kS + kv0 + r)      * kDH + t * 32 + kb);
      bf16x16 bk1 = load_fragB(Kk + ((size_t)bh * kS + kv0 + 16 + r) * kDH + t * 32 + kb);
      s0 = __builtin_amdgcn_wmma_f32_16x16x32_bf16(false, aq[t], false, bk0, (short)0, s0, false, false);
      s1 = __builtin_amdgcn_wmma_f32_16x16x32_bf16(false, aq[t], false, bk1, (short)0, s1, false, false);
    }

    // ---- scale + bias (log2 domain), online softmax ----
    // bias is 256MB of single-use data: stream it non-temporally past L2.
    const float* bp = bias + ((size_t)h * kS + q0 + 8 * hl) * kS + kv0 + nn;
    float nm[8], corr[8];
#pragma unroll
    for (int j = 0; j < 8; ++j) {
      float b0 = __builtin_nontemporal_load(bp + (size_t)j * kS);
      float b1 = __builtin_nontemporal_load(bp + (size_t)j * kS + 16);
      s0[j] = s0[j] * sc + b0 * bl;
      s1[j] = s1[j] * sc + b1 * bl;
      float mt = redmax16(fmaxf(s0[j], s1[j]));
      nm[j]   = fmaxf(rm[j], mt);
      corr[j] = exp2f(rm[j] - nm[j]);
      rm[j]   = nm[j];
    }
#pragma unroll
    for (int j = 0; j < 8; ++j) {
      s0[j] = exp2f(s0[j] - nm[j]);
      s1[j] = exp2f(s1[j] - nm[j]);
      float rs = redsum16(s0[j] + s1[j]);
      rl[j] = rl[j] * corr[j] + rs;
    }
#pragma unroll
    for (int f = 0; f < 4; ++f)
#pragma unroll
      for (int j = 0; j < 8; ++j) acc[f][j] *= corr[j];

    // ---- P: C-layout -> LDS row-major 16x32 (bf16) ----
#pragma unroll
    for (int j = 0; j < 8; ++j) {
      pw[(j + 8 * hl) * LDP + nn]      = f32_to_bf16_rne(s0[j]);
      pw[(j + 8 * hl) * LDP + nn + 16] = f32_to_bf16_rne(s1[j]);
    }
    __syncthreads();

    // ---- O += P * V ----
    bf16x16 ap = load_fragA(pw + r * LDP + kf);
#pragma unroll
    for (int f = 0; f < 4; ++f) {
      bf16x16 bv = load_fragB(Vt + ((size_t)bh * kDH + f * 16 + r) * kS + kv0 + kb);
      acc[f] = __builtin_amdgcn_wmma_f32_16x16x32_bf16(false, ap, false, bv, (short)0, acc[f], false, false);
    }
    __syncthreads();
  }

  // ---- normalize + write ctx[B,S,D] bf16 ----
  float inv[8];
#pragma unroll
  for (int j = 0; j < 8; ++j) inv[j] = 1.0f / rl[j];
#pragma unroll
  for (int f = 0; f < 4; ++f)
#pragma unroll
    for (int j = 0; j < 8; ++j) {
      int qrow = q0 + j + 8 * hl;
      int dh = f * 16 + nn;
      ctx[((size_t)(b * kS + qrow)) * kD + h * kDH + dh] = f32_to_bf16_rne(acc[f][j] * inv[j]);
    }
}

}  // namespace

extern "C" void kernel_launch(void* const* d_in, const int* in_sizes, int n_in,
                              void* d_out, int out_size, void* d_ws, size_t ws_size,
                              hipStream_t stream) {
  (void)in_sizes; (void)n_in; (void)out_size; (void)ws_size;
  const float* X    = (const float*)d_in[0];
  const float* bias = (const float*)d_in[1];
  const float* Wq   = (const float*)d_in[2];
  const float* Wk   = (const float*)d_in[3];
  const float* Wv   = (const float*)d_in[4];
  const float* Wo   = (const float*)d_in[5];

  char* ws = (char*)d_ws;
  unsigned short* Xb  = (unsigned short*)(ws);                        // 8 MiB
  unsigned short* Wqb = (unsigned short*)(ws + (size_t)( 8u << 20));  // 2 MiB each
  unsigned short* Wkb = (unsigned short*)(ws + (size_t)(10u << 20));
  unsigned short* Wvb = (unsigned short*)(ws + (size_t)(12u << 20));
  unsigned short* Wob = (unsigned short*)(ws + (size_t)(14u << 20));
  unsigned short* Qb  = (unsigned short*)(ws + (size_t)(16u << 20));  // [B,H,S,DH]
  unsigned short* Kb  = (unsigned short*)(ws + (size_t)(24u << 20));  // [B,H,S,DH]
  unsigned short* Vtb = (unsigned short*)(ws + (size_t)(32u << 20));  // [B,H,DH,S]
  unsigned short* Ctx = (unsigned short*)(ws + (size_t)(40u << 20));  // [B*S, D]

  const int nX = kM * kD;   // 4194304
  const int nW = kD * kD;   // 1048576
  cvt_f32_to_bf16<<<(nX / 4 + 255) / 256, 256, 0, stream>>>(X,  Xb,  nX / 4);
  cvt_f32_to_bf16<<<(nW / 4 + 255) / 256, 256, 0, stream>>>(Wq, Wqb, nW / 4);
  cvt_f32_to_bf16<<<(nW / 4 + 255) / 256, 256, 0, stream>>>(Wk, Wkb, nW / 4);
  cvt_f32_to_bf16<<<(nW / 4 + 255) / 256, 256, 0, stream>>>(Wv, Wvb, nW / 4);
  cvt_f32_to_bf16<<<(nW / 4 + 255) / 256, 256, 0, stream>>>(Wo, Wob, nW / 4);

  dim3 gg(kD / 128, kM / 128);  // (8, 32)
  gemm_bf16_wmma<1><<<gg, 256, 0, stream>>>(Xb, Wqb, Qb,  kM, kD, kD);
  gemm_bf16_wmma<1><<<gg, 256, 0, stream>>>(Xb, Wkb, Kb,  kM, kD, kD);
  gemm_bf16_wmma<2><<<gg, 256, 0, stream>>>(Xb, Wvb, Vtb, kM, kD, kD);

  attn_fwd<<<dim3(kS / 128, kH, kB), 256, 0, stream>>>(Qb, Kb, Vtb, bias, Ctx);

  gemm_bf16_wmma<0><<<gg, 256, 0, stream>>>(Ctx, Wob, d_out, kM, kD, kD);
}
